// Attention_3994319585930
// MI455X (gfx1250) — matrix operand order repaired and checked by reading, not compiled, not run on hardware
//
#include <hip/hip_runtime.h>
#include <hip/hip_bf16.h>
#include <math.h>

typedef _Float16 v16h __attribute__((ext_vector_type(16)));
typedef _Float16 v8h  __attribute__((ext_vector_type(8)));
typedef float    v8f  __attribute__((ext_vector_type(8)));
typedef unsigned int u32x4 __attribute__((ext_vector_type(4)));
typedef int          i32x4 __attribute__((ext_vector_type(4)));
typedef int          i32x8 __attribute__((ext_vector_type(8)));

union FragH { v16h v; v8h h8[2]; };

#define BATCH 2
#define SEQ   4096
#define DIM   512
#define NHEAD 8
#define HDIM  64
#define NCH   1024
#define FCH   4
#define ROTD  32

#if __has_builtin(__builtin_amdgcn_tensor_load_to_lds) && \
    __has_builtin(__builtin_amdgcn_s_wait_tensorcnt)
#define USE_TDM 1
#endif

// ---------------------------------------------------------------------------
// TDM 2D tile load: global (row-major, stride_elems between rows, f16 data)
// -> LDS at lds_byte_addr, tile tile_w x tile_h elements, rows packed
// contiguously (tile_w elements per LDS row). D# layout per ISA ch.8.
// ---------------------------------------------------------------------------
__device__ __forceinline__ void tdm_load_2d_f16(unsigned lds_byte_addr,
                                                const void* gptr,
                                                unsigned tile_w, unsigned tile_h,
                                                unsigned stride_elems)
{
#ifdef USE_TDM
    unsigned long long ga = (unsigned long long)(size_t)gptr;
    u32x4 g0;
    g0[0] = 1u;                                        // count=1, user D#
    g0[1] = lds_byte_addr;                             // lds_addr (bytes)
    g0[2] = (unsigned)(ga & 0xFFFFFFFFu);              // global_addr[31:0]
    g0[3] = (unsigned)((ga >> 32) & 0x1FFFFFFu)        // global_addr[56:32]
          | (2u << 30);                                // type=2 ("image")
    i32x8 g1;
    g1[0] = (int)(1u << 16);                           // wg_mask=0, data_size=1 (2B)
    g1[1] = (int)((tile_w & 0xFFFFu) << 16);           // tensor_dim0[15:0] = tile_w
    g1[2] = (int)(((tile_w >> 16) & 0xFFFFu)           // tensor_dim0[31:16]
          | ((tile_h & 0xFFFFu) << 16));               // tensor_dim1[15:0] = tile_h
    g1[3] = (int)(((tile_h >> 16) & 0xFFFFu)           // tensor_dim1[31:16]
          | ((tile_w & 0xFFFFu) << 16));               // tile_dim0 = tile_w
    g1[4] = (int)(tile_h & 0xFFFFu);                   // tile_dim1 = tile_h, tile_dim2=0
    g1[5] = (int)stride_elems;                         // tensor_dim0_stride[31:0]
    g1[6] = 0;                                         // stride0[47:32], stride1[15:0]
    g1[7] = 0;                                         // stride1[47:16]
    i32x4 z4 = {0, 0, 0, 0};
#if defined(__clang_major__) && __clang_major__ >= 23
    i32x8 z8 = {0, 0, 0, 0, 0, 0, 0, 0};
    __builtin_amdgcn_tensor_load_to_lds(g0, g1, z4, z4, z8, 0);
#else
    __builtin_amdgcn_tensor_load_to_lds(g0, g1, z4, z4, 0);
#endif
#else
    (void)lds_byte_addr; (void)gptr; (void)tile_w; (void)tile_h; (void)stride_elems;
#endif
}

// ---------------------------------------------------------------------------
// Kernel 1: cast x -> f16, transpose+cast W_qkv (512x1536 -> 1536x512) and
// W_out (512x512 -> 512x512) so GEMM B-fragments are contiguous in K.
// ---------------------------------------------------------------------------
__global__ __launch_bounds__(256)
void pack_kernel(const float* __restrict__ x, const float* __restrict__ Wqkv,
                 const float* __restrict__ Wout,
                 _Float16* __restrict__ xh, _Float16* __restrict__ wqkvT,
                 _Float16* __restrict__ woutT)
{
    const size_t stride = (size_t)gridDim.x * 256u;
    const size_t i0 = (size_t)blockIdx.x * 256u + threadIdx.x;

    for (size_t i = i0; i < (size_t)BATCH * SEQ * DIM; i += stride)
        xh[i] = (_Float16)x[i];

    for (size_t i = i0; i < (size_t)(3 * NHEAD * HDIM) * DIM; i += stride) {
        size_t n = i / DIM, k = i % DIM;                 // wqkvT[n][k]
        wqkvT[i] = (_Float16)Wqkv[k * (3 * NHEAD * HDIM) + n];
    }

    for (size_t i = i0; i < (size_t)DIM * DIM; i += stride) {
        size_t n = i / DIM, k = i % DIM;                 // woutT[n][k]
        woutT[i] = (_Float16)Wout[k * DIM + n];
    }
}

// ---------------------------------------------------------------------------
// f16 WMMA GEMM with TDM double-buffered LDS staging.
// C[MxN] = A[MxK] * Bt[NxK]^T (+bias). Block = 64x64 tile, 8 waves, each
// wave computes 16x32 (one A-fragment feeding two WMMAs per K-step).
// Per K-step of 32: TDM loads A-tile (64x32) and B-tile (64x32) into LDS
// for the *next* step while WMMAs consume the current buffers.
// ---------------------------------------------------------------------------
template<bool OUT16, bool BIAS>
__global__ __launch_bounds__(256)
void gemm_wmma_kernel(const _Float16* __restrict__ A, const _Float16* __restrict__ Bt,
                      _Float16* __restrict__ C16, float* __restrict__ C32,
                      const float* __restrict__ bias, int M, int N, int K)
{
    const int tid  = threadIdx.x;
    const int wave = tid >> 5;
    const int lane = tid & 31;
    const int c    = lane & 15;           // col (B/C/D) or row (A) within half
    const int kh   = lane >> 4;           // which K-half this lane holds
    const int wm   = wave >> 1;           // 0..3  -> 16-row strip
    const int wn   = wave & 1;            // 0..1  -> 32-col strip

    const int bm0 = blockIdx.y * 64;
    const int bn0 = blockIdx.x * 64;

    __shared__ __align__(32) _Float16 sA[2][64 * 32];
    __shared__ __align__(32) _Float16 sB[2][64 * 32];

    v8f acc0, acc1;
    for (int e = 0; e < 8; ++e) { acc0[e] = 0.0f; acc1[e] = 0.0f; }

    const int NT = K / 32;

#ifdef USE_TDM
    if (wave == 0) {
        tdm_load_2d_f16((unsigned)(size_t)&sA[0][0], A  + (size_t)bm0 * K, 32, 64, K);
        tdm_load_2d_f16((unsigned)(size_t)&sB[0][0], Bt + (size_t)bn0 * K, 32, 64, K);
    }
#endif

    for (int it = 0; it < NT; ++it) {
        const int cur = it & 1;
#ifdef USE_TDM
        if (wave == 0) __builtin_amdgcn_s_wait_tensorcnt(0);
        __syncthreads();                       // tile `it` visible to all waves
        if (wave == 0 && it + 1 < NT) {
            const int kk = (it + 1) * 32;
            tdm_load_2d_f16((unsigned)(size_t)&sA[cur ^ 1][0],
                            A + (size_t)bm0 * K + kk, 32, 64, K);
            tdm_load_2d_f16((unsigned)(size_t)&sB[cur ^ 1][0],
                            Bt + (size_t)bn0 * K + kk, 32, 64, K);
        }
#else
        __syncthreads();                       // previous consumers done
        {   // cooperative staging fallback: 2 halves per u32, 16 u32 per row
            const int kk = it * 32;
            const unsigned* gA = (const unsigned*)(A  + (size_t)bm0 * K + kk);
            const unsigned* gB = (const unsigned*)(Bt + (size_t)bn0 * K + kk);
            unsigned* lA = (unsigned*)&sA[cur][0];
            unsigned* lB = (unsigned*)&sB[cur][0];
            const int rowW = K >> 1;           // u32 per global row
            for (int e = tid; e < 64 * 16; e += 256) {
                const int r = e >> 4, q = e & 15;
                lA[r * 16 + q] = gA[(size_t)r * rowW + q];
                lB[r * 16 + q] = gB[(size_t)r * rowW + q];
            }
        }
        __syncthreads();
#endif
        const _Float16* aRow = &sA[cur][(wm * 16 + c) * 32];
        FragH a, b0, b1;
        a.h8[0] = *(const v8h*)(aRow + 16 * kh);
        a.h8[1] = *(const v8h*)(aRow + 16 * kh + 8);
        b0.v = *(const v16h*)(&sB[cur][(wn * 32 + c) * 32 + 16 * kh]);
        b1.v = *(const v16h*)(&sB[cur][(wn * 32 + 16 + c) * 32 + 16 * kh]);
        acc0 = __builtin_amdgcn_wmma_f32_16x16x32_f16(false, a.v, false, b0.v,
                                                      (short)0, acc0, false, false);
        acc1 = __builtin_amdgcn_wmma_f32_16x16x32_f16(false, a.v, false, b1.v,
                                                      (short)0, acc1, false, false);
    }

    const int col0 = bn0 + wn * 32 + c;
    const int col1 = col0 + 16;
    const float bb0 = BIAS ? bias[col0] : 0.0f;
    const float bb1 = BIAS ? bias[col1] : 0.0f;
    for (int e = 0; e < 8; ++e) {
        const int row = bm0 + wm * 16 + e + 8 * kh;
        const float v0 = acc0[e] + bb0;
        const float v1 = acc1[e] + bb1;
        if (OUT16) {
            C16[(size_t)row * N + col0] = (_Float16)v0;
            C16[(size_t)row * N + col1] = (_Float16)v1;
        } else {
            C32[(size_t)row * N + col0] = v0;
            C32[(size_t)row * N + col1] = v1;
        }
    }
}

// ---------------------------------------------------------------------------
// Kernel 3: RoPE (first 32 dims) + q scale, scatter qkv (8192x1536 f16)
// into head-major Q/K/V: [((b*8+h)*4+f)*1024 + n][64] f16.
// ---------------------------------------------------------------------------
__global__ __launch_bounds__(256)
void rope_kernel(const _Float16* __restrict__ qkv, const float* __restrict__ sinT,
                 const float* __restrict__ cosT,
                 _Float16* __restrict__ Qh, _Float16* __restrict__ Kh,
                 _Float16* __restrict__ Vh)
{
    const size_t idx = (size_t)blockIdx.x * 256u + threadIdx.x;  // 4194304 total
    const int    ci  = (int)(idx & 511);
    const size_t row = idx >> 9;                // 0..8191
    const int h = ci >> 6;
    const int d = ci & 63;
    const int b = (int)(row >> 12);
    const int t = (int)(row & 4095);
    const int f = t >> 10;
    const int n = t & 1023;

    const _Float16* base = qkv + row * (3 * DIM);
    float qv = (float)base[h * HDIM + d];
    float kv = (float)base[DIM + h * HDIM + d];
    float vv = (float)base[2 * DIM + h * HDIM + d];

    if (d < ROTD) {
        const float s  = sinT[n * ROTD + d];
        const float cc = cosT[n * ROTD + d];
        const int   pd = d ^ 1;
        const float qp = (float)base[h * HDIM + pd];
        const float kp = (float)base[DIM + h * HDIM + pd];
        const float sgn = (d & 1) ? 1.0f : -1.0f;   // rotate_every_two
        qv = qv * cc + sgn * qp * s;
        kv = kv * cc + sgn * kp * s;
    }
    qv *= 0.125f;   // DH^-0.5

    const size_t dst = ((((size_t)b * NHEAD + h) * FCH + f) * NCH + n) * HDIM + d;
    Qh[dst] = (_Float16)qv;
    Kh[dst] = (_Float16)kv;
    Vh[dst] = (_Float16)vv;
}

// ---------------------------------------------------------------------------
// Kernel 4: flash attention per (b,h,f). Block = 4 waves x 16 query rows.
// Keys processed in tiles of 32 with online softmax; all matmuls via WMMA.
// ---------------------------------------------------------------------------
__global__ __launch_bounds__(128)
void attn_kernel(const _Float16* __restrict__ Qg, const _Float16* __restrict__ Kg,
                 const _Float16* __restrict__ Vg, _Float16* __restrict__ Og)
{
    const int tid  = threadIdx.x;
    const int wave = tid >> 5;            // 0..3
    const int lane = tid & 31;
    const int c    = lane & 15;
    const int kh   = lane >> 4;

    const int combo = blockIdx.x;         // 0..63 = (b, h, f)
    const int b = combo >> 5;
    const int h = (combo >> 2) & 7;
    const int f = combo & 3;
    const int q0 = blockIdx.y * 64 + wave * 16;

    const size_t head = (((size_t)b * NHEAD + h) * FCH + f) * NCH;
    const _Float16* Qp = Qg + head * HDIM;
    const _Float16* Kp = Kg + head * HDIM;
    const _Float16* Vp = Vg + head * HDIM;

    __shared__ __align__(32) _Float16 sVt[HDIM * 32];     // V tile transposed: [d][key]
    __shared__ __align__(16) float    sS[4][16 * 32];     // per-wave raw scores
    __shared__ __align__(16) _Float16 sP[4][16 * 32];     // per-wave exp probs
    __shared__ float sScale[4][16];
    __shared__ float sL[4][16];

    FragH qf[2];
    {
        const _Float16* qrow = Qp + (size_t)(q0 + c) * HDIM;
        qf[0].h8[0] = *(const v8h*)(qrow + 0  + 16 * kh);
        qf[0].h8[1] = *(const v8h*)(qrow + 8  + 16 * kh);
        qf[1].h8[0] = *(const v8h*)(qrow + 32 + 16 * kh);
        qf[1].h8[1] = *(const v8h*)(qrow + 40 + 16 * kh);
    }

    v8f acc[4];
    for (int t = 0; t < 4; ++t)
        for (int e = 0; e < 8; ++e) acc[t][e] = 0.0f;

    float mrow = -INFINITY, lrow = 0.0f;   // valid in lanes 0..15

    for (int j = 0; j < NCH; j += 32) {
        __syncthreads();   // previous iteration's sVt/sP consumers done

        // Stage V tile transposed: coalesced v8h global loads, 8-way LDS scatter.
        // 32 keys x 64 dims = 256 v8h chunks; 2 per thread.
        for (int e = tid; e < 32 * HDIM / 8; e += 128) {
            const int key = e >> 3;              // 8 chunks per key row
            const int d0  = (e & 7) * 8;
            const v8h vv = *(const v8h*)(Vp + (size_t)(j + key) * HDIM + d0);
            for (int q = 0; q < 8; ++q)
                sVt[(d0 + q) * 32 + key] = vv[q];
        }

        // Prefetch next K tile a full iteration ahead (global_prefetch_b8)
        if (j + 32 < NCH)
            __builtin_prefetch(Kp + (size_t)(j + 32 + lane) * HDIM, 0, 0);

        v8f sa, sb;
        for (int e = 0; e < 8; ++e) { sa[e] = 0.0f; sb[e] = 0.0f; }
        for (int s = 0; s < 2; ++s) {   // d-step kk = 32*s
            FragH ka, kb;
            ka.v = *(const v16h*)(Kp + (size_t)(j + c) * HDIM      + 32 * s + 16 * kh);
            kb.v = *(const v16h*)(Kp + (size_t)(j + 16 + c) * HDIM + 32 * s + 16 * kh);
            sa = __builtin_amdgcn_wmma_f32_16x16x32_f16(false, qf[s].v, false, ka.v,
                                                        (short)0, sa, false, false);
            sb = __builtin_amdgcn_wmma_f32_16x16x32_f16(false, qf[s].v, false, kb.v,
                                                        (short)0, sb, false, false);
        }

        float* Sw = sS[wave];
        for (int e = 0; e < 8; ++e) {
            Sw[(e + 8 * kh) * 32 + c]      = sa[e];
            Sw[(e + 8 * kh) * 32 + 16 + c] = sb[e];
        }
        __syncthreads();

        if (lane < 16) {
            const float* srow = &sS[wave][lane * 32];
            float mt = srow[0];
            for (int e = 1; e < 32; ++e) mt = fmaxf(mt, srow[e]);
            const float mnew = fmaxf(mrow, mt);
            const float sc   = __expf(mrow - mnew);
            float sum = 0.0f;
            _Float16* prow = &sP[wave][lane * 32];
            for (int e = 0; e < 32; ++e) {
                const float p = __expf(srow[e] - mnew);
                sum += p;
                prow[e] = (_Float16)p;
            }
            lrow = lrow * sc + sum;
            mrow = mnew;
            sScale[wave][lane] = sc;
        }
        __syncthreads();

        float scv[8];
        for (int e = 0; e < 8; ++e) scv[e] = sScale[wave][e + 8 * kh];
        for (int t = 0; t < 4; ++t)
            for (int e = 0; e < 8; ++e) acc[t][e] *= scv[e];

        FragH pf;
        pf.h8[0] = *(const v8h*)(&sP[wave][c * 32 + 16 * kh]);
        pf.h8[1] = *(const v8h*)(&sP[wave][c * 32 + 8 + 16 * kh]);
        for (int t = 0; t < 4; ++t) {
            FragH vf;
            vf.v = *(const v16h*)(&sVt[(t * 16 + c) * 32 + 16 * kh]);
            acc[t] = __builtin_amdgcn_wmma_f32_16x16x32_f16(false, pf.v, false, vf.v,
                                                            (short)0, acc[t], false, false);
        }
    }

    __syncthreads();
    if (lane < 16) sL[wave][lane] = lrow;
    __syncthreads();

    float rinv[8];
    for (int e = 0; e < 8; ++e) rinv[e] = 1.0f / sL[wave][e + 8 * kh];

    for (int t = 0; t < 4; ++t) {
        const int d = t * 16 + c;
        for (int e = 0; e < 8; ++e) {
            const int row = q0 + e + 8 * kh;
            const size_t off = ((size_t)(b * SEQ + f * NCH + row)) * DIM + h * HDIM + d;
            Og[off] = (_Float16)(acc[t][e] * rinv[e]);
        }
    }
}

// ---------------------------------------------------------------------------
// Host launcher
// ---------------------------------------------------------------------------
extern "C" void kernel_launch(void* const* d_in, const int* in_sizes, int n_in,
                              void* d_out, int out_size, void* d_ws, size_t ws_size,
                              hipStream_t stream) {
    const float* x    = (const float*)d_in[0];
    const float* Wqkv = (const float*)d_in[1];
    const float* Wout = (const float*)d_in[2];
    const float* bout = (const float*)d_in[3];
    const float* sinT = (const float*)d_in[4];
    const float* cosT = (const float*)d_in[5];
    (void)in_sizes; (void)n_in; (void)out_size; (void)ws_size;

    char* ws = (char*)d_ws;
    const size_t ROWS = (size_t)BATCH * SEQ;                  // 8192
    size_t off = 0;
    _Float16* xh    = (_Float16*)(ws + off); off += ROWS * DIM * 2;                        // 8 MB
    _Float16* wqkvT = (_Float16*)(ws + off); off += (size_t)(3 * NHEAD * HDIM) * DIM * 2;  // 1.5 MB
    _Float16* woutT = (_Float16*)(ws + off); off += (size_t)DIM * DIM * 2;                 // 0.5 MB
    _Float16* qkvh  = (_Float16*)(ws + off); off += ROWS * (size_t)(3 * DIM) * 2;          // 24 MB
    _Float16* Qh    = (_Float16*)(ws + off); off += ROWS * DIM * 2;
    _Float16* Kh    = (_Float16*)(ws + off); off += ROWS * DIM * 2;
    _Float16* Vh    = (_Float16*)(ws + off); off += ROWS * DIM * 2;
    _Float16* attnH = (_Float16*)(ws + off); off += ROWS * DIM * 2;

    // 1. pack / cast / transpose weights
    pack_kernel<<<2048, 256, 0, stream>>>(x, Wqkv, Wout, xh, wqkvT, woutT);

    // 2. QKV projection: (8192x512) x (512x1536) -> f16
    gemm_wmma_kernel<true, false>
        <<<dim3((3 * NHEAD * HDIM) / 64, (int)(ROWS / 64)), 256, 0, stream>>>(
            xh, wqkvT, qkvh, nullptr, nullptr, (int)ROWS, 3 * NHEAD * HDIM, DIM);

    // 3. RoPE + scale + head-major scatter
    rope_kernel<<<(int)(ROWS * DIM / 256), 256, 0, stream>>>(qkvh, sinT, cosT, Qh, Kh, Vh);

    // 4. chunked flash attention: 64 (b,h,f) combos x 16 query tiles of 64 rows
    attn_kernel<<<dim3(64, NCH / 64), 128, 0, stream>>>(Qh, Kh, Vh, attnH);

    // 5. output projection + bias -> f32 d_out
    gemm_wmma_kernel<false, true>
        <<<dim3(DIM / 64, (int)(ROWS / 64)), 256, 0, stream>>>(
            attnH, woutT, nullptr, (float*)d_out, bout, (int)ROWS, DIM, DIM);
}
